// TopKMoE_Fast_52673478918146
// MI455X (gfx1250) — matrix operand order repaired and checked by reading, not compiled, not run on hardware
//
#include <hip/hip_runtime.h>
#include <hip/hip_bf16.h>

// ---------------- problem constants (from reference) ----------------
constexpr int BT   = 8192;          // 4*2048 tokens
constexpr int DIM  = 1024;          // D
constexpr int HID  = 4096;          // H
constexpr int NEXP = 8;             // experts
constexpr int TOPK = 2;
constexpr int BTK  = BT * TOPK;     // 16384 assignment slots
constexpr int CAP  = 2560;          // ceil(1.25 * BTK / NEXP)
constexpr int KT   = 32;            // WMMA K step (bf16 16x16x32)
constexpr int LDSS = KT + 8;        // padded LDS row stride (ushorts)

typedef __attribute__((ext_vector_type(16))) __bf16 v16bf;
typedef __attribute__((ext_vector_type(8)))  float  v8f;

union Frag { v16bf v; unsigned int u[8]; };

// packed f32x2 -> bf16x2 (lowers to v_cvt_pk_bf16_f32 where supported)
__device__ __forceinline__ unsigned int pack_bf16(float a, float b) {
  __hip_bfloat162 h = __float22bfloat162_rn(make_float2(a, b));
  unsigned int u;
  __builtin_memcpy(&u, &h, sizeof(u));
  return u;
}

__device__ __forceinline__ unsigned short f2bf(float f) {
  unsigned u = __builtin_bit_cast(unsigned, f);
  u += 0x7FFFu + ((u >> 16) & 1u);            // round-to-nearest-even
  return (unsigned short)(u >> 16);
}

__device__ __forceinline__ float gelu_tanh(float v) {
  float c = v + 0.044715f * v * v * v;
  return 0.5f * v * (1.0f + tanhf(0.7978845608028654f * c));
}

// ---------------- utility: zero a float buffer ----------------
__global__ void k_zero(float* __restrict__ p, int n) {
  int i = blockIdx.x * 256 + threadIdx.x;
  if (i < n) p[i] = 0.0f;
}

// ---------------- gating: logits, softmax, top-2, renorm ----------------
__global__ __launch_bounds__(256) void k_gate(const float* __restrict__ x,
                                              const float* __restrict__ Wg,
                                              int* __restrict__ assign_e,
                                              float* __restrict__ assign_w,
                                              int* __restrict__ counts) {
  int t    = blockIdx.x * 8 + (threadIdx.x >> 5);   // one wave32 per token
  int lane = threadIdx.x & 31;
  const float* xr = x + (size_t)t * DIM;

  float acc[NEXP];
#pragma unroll
  for (int n = 0; n < NEXP; ++n) acc[n] = 0.0f;

  for (int j = lane; j < DIM; j += 32) {
    float xv = xr[j];
#pragma unroll
    for (int n = 0; n < NEXP; ++n) acc[n] += xv * Wg[n * DIM + j];
  }
#pragma unroll
  for (int m = 16; m > 0; m >>= 1) {
#pragma unroll
    for (int n = 0; n < NEXP; ++n) acc[n] += __shfl_xor(acc[n], m, 32);
  }

  if (lane == 0) {
    float mx = acc[0];
#pragma unroll
    for (int n = 1; n < NEXP; ++n) mx = fmaxf(mx, acc[n]);
    float p[NEXP], s = 0.0f;
#pragma unroll
    for (int n = 0; n < NEXP; ++n) { p[n] = __expf(acc[n] - mx); s += p[n]; }
    float inv = 1.0f / s;
#pragma unroll
    for (int n = 0; n < NEXP; ++n) p[n] *= inv;

    // stable top-2 (ties -> lowest index), matching jax.lax.top_k
    int i1 = 0;
#pragma unroll
    for (int n = 1; n < NEXP; ++n) if (p[n] > p[i1]) i1 = n;
    int i2 = (i1 == 0) ? 1 : 0;
#pragma unroll
    for (int n = 0; n < NEXP; ++n)
      if (n != i1 && p[n] > p[i2]) i2 = n;

    float v1 = p[i1], v2 = p[i2];
    float denom = fmaxf(v1 + v2, 1e-9f);
    assign_e[2 * t]     = i1;
    assign_w[2 * t]     = v1 / denom;
    assign_e[2 * t + 1] = i2;
    assign_w[2 * t + 1] = v2 / denom;
    atomicAdd(&counts[i1], 1);
    atomicAdd(&counts[i2], 1);
  }
}

// ---------------- deterministic capacity selection (rank-based top-C) ----------------
__global__ __launch_bounds__(1024) void k_select(const int* __restrict__ assign_e,
                                                 const float* __restrict__ assign_w,
                                                 int* __restrict__ tok,
                                                 float* __restrict__ w_eff) {
  int e = blockIdx.x;
  for (int i = threadIdx.x; i < BTK; i += 1024) {
    if (assign_e[i] != e) continue;
    float wi = assign_w[i];
    int rank = 0;
    for (int j = 0; j < BTK; ++j) {
      if (assign_e[j] == e) {
        float wj = assign_w[j];
        rank += (wj > wi) || (wj == wi && j < i);
      }
    }
    if (rank < CAP) {
      tok[e * CAP + rank]   = i >> 1;   // token id = slot / k
      w_eff[e * CAP + rank] = wi;
    }
  }
}

// ---------------- lb_loss and overflow ----------------
__global__ void k_losses(const int* __restrict__ counts,
                         float* __restrict__ out_loss,
                         float* __restrict__ out_ovf) {
  if (threadIdx.x == 0 && blockIdx.x == 0) {
    const float expected = (float)BTK / NEXP;     // 2048
    float s = 0.0f; int kept = 0;
    for (int n = 0; n < NEXP; ++n) {
      float d = (float)counts[n] - expected;
      s += d * d;
      int c = counts[n]; kept += (c < CAP) ? c : CAP;
    }
    *out_loss = (s / NEXP) / (expected * expected);
    *out_ovf  = (float)(BTK - kept) / (float)BTK;
  }
}

// ---------------- GEMM1: h = GELU(gather(x)[CAP,DIM] @ W1[e] + b1), bf16 out ----------------
__global__ __launch_bounds__(256) void k_ffn1(const float* __restrict__ x,
                                              const float* __restrict__ W1,
                                              const float* __restrict__ b1,
                                              const int* __restrict__ tok,
                                              unsigned short* __restrict__ hout,
                                              int e) {
  __shared__ unsigned short As[128][LDSS];   // gathered x tile, bf16, [M][K]
  __shared__ unsigned short Bs[128][LDSS];   // W1 tile transposed, bf16, [N][K]
  __shared__ int toks[128];

  const int tid  = threadIdx.x;
  const int wave = tid >> 5, lane = tid & 31;
  const int half = lane >> 4, r = lane & 15;
  const int m0 = blockIdx.y * 128, n0 = blockIdx.x * 128;

  if (tid < 128) toks[tid] = tok[e * CAP + m0 + tid];
  __syncthreads();

  const float* W1e = W1 + (size_t)e * DIM * HID;
  v8f acc[8];
  const v8f vzero = {0.f, 0.f, 0.f, 0.f, 0.f, 0.f, 0.f, 0.f};
#pragma unroll
  for (int i = 0; i < 8; ++i) acc[i] = vzero;

  for (int k0 = 0; k0 < DIM; k0 += KT) {
    // ---- stage A (gathered token rows), f32 -> bf16, packed dword stores ----
#pragma unroll
    for (int it = 0; it < 4; ++it) {
      int q   = tid + it * 256;        // 1024 float4 chunks = 128 rows x 8
      int row = q >> 3;
      int kq  = (q & 7) * 4;
      const float4 f4 = *(const float4*)(x + (size_t)toks[row] * DIM + k0 + kq);
      *(unsigned int*)&As[row][kq + 0] = pack_bf16(f4.x, f4.y);
      *(unsigned int*)&As[row][kq + 2] = pack_bf16(f4.z, f4.w);
    }
    // ---- stage B (W1 tile, transposed to [N][K]); two K rows per thread so
    //      bf16 pairs along K pack into dword LDS stores ----
#pragma unroll
    for (int it = 0; it < 2; ++it) {
      int q  = tid + it * 256;         // 512 chunks = 16 K-pairs x 32 N-chunks
      int kp = q >> 5;                 // K pair index (0..15)
      int nq = (q & 31) * 4;
      const float* src = W1e + (size_t)(k0 + 2 * kp) * HID + n0 + nq;
      const float4 fa = *(const float4*)src;
      const float4 fb = *(const float4*)(src + HID);
      if (k0 + KT < DIM) __builtin_prefetch(src + (size_t)KT * HID, 0, 0);
      *(unsigned int*)&Bs[nq + 0][2 * kp] = pack_bf16(fa.x, fb.x);
      *(unsigned int*)&Bs[nq + 1][2 * kp] = pack_bf16(fa.y, fb.y);
      *(unsigned int*)&Bs[nq + 2][2 * kp] = pack_bf16(fa.z, fb.z);
      *(unsigned int*)&Bs[nq + 3][2 * kp] = pack_bf16(fa.w, fb.w);
    }
    __syncthreads();

    // A fragment: 16x32 bf16 per ISA layout (lane&15 = row; K groups by half)
    Frag a;
    int arow = wave * 16 + r;
#pragma unroll
    for (int v = 0; v < 8; ++v) {
      int k = (v & 3) * 2 + (v >> 2) * 16 + half * 8;
      a.u[v] = *(const unsigned int*)&As[arow][k];
    }
    // software-pipelined B fragments: load next while WMMA on current
    Frag bcur, bnxt;
#pragma unroll
    for (int v = 0; v < 8; ++v)
      bcur.u[v] = *(const unsigned int*)&Bs[r][(v & 3) * 2 + (v >> 2) * 16 + half * 8];
#pragma unroll
    for (int nt = 0; nt < 8; ++nt) {
      if (nt < 7) {
        int bcol = (nt + 1) * 16 + r;
#pragma unroll
        for (int v = 0; v < 8; ++v)
          bnxt.u[v] = *(const unsigned int*)&Bs[bcol][(v & 3) * 2 + (v >> 2) * 16 + half * 8];
      }
      acc[nt] = __builtin_amdgcn_wmma_f32_16x16x32_bf16(
          false, a.v, false, bcur.v, (short)0, acc[nt], false, false);
      bcur = bnxt;
    }
    __syncthreads();
  }

  // epilogue: +b1, GELU, bf16 store
#pragma unroll
  for (int nt = 0; nt < 8; ++nt) {
#pragma unroll
    for (int v = 0; v < 8; ++v) {
      int row = m0 + wave * 16 + half * 8 + v;
      int col = n0 + nt * 16 + r;
      float val = acc[nt][v] + b1[e * HID + col];
      hout[(size_t)row * HID + col] = f2bf(gelu_tanh(val));
    }
  }
}

// ---------------- GEMM2: y[tok] += w * (h @ W2[e] + b2) ----------------
__global__ __launch_bounds__(256) void k_ffn2(const unsigned short* __restrict__ h,
                                              const float* __restrict__ W2,
                                              const float* __restrict__ b2,
                                              const int* __restrict__ tok,
                                              const float* __restrict__ w_eff,
                                              float* __restrict__ y,
                                              int e) {
  __shared__ unsigned short As[128][LDSS];
  __shared__ unsigned short Bs[128][LDSS];

  const int tid  = threadIdx.x;
  const int wave = tid >> 5, lane = tid & 31;
  const int half = lane >> 4, r = lane & 15;
  const int m0 = blockIdx.y * 128, n0 = blockIdx.x * 128;

  const float* W2e = W2 + (size_t)e * HID * DIM;
  v8f acc[8];
  const v8f vzero = {0.f, 0.f, 0.f, 0.f, 0.f, 0.f, 0.f, 0.f};
#pragma unroll
  for (int i = 0; i < 8; ++i) acc[i] = vzero;

  for (int k0 = 0; k0 < HID; k0 += KT) {
    // ---- stage A (h tile, already bf16): 2048 dword chunks = 128 rows x 16 ----
#pragma unroll
    for (int it = 0; it < 8; ++it) {
      int q   = tid + it * 256;
      int row = q >> 4;
      int ku  = (q & 15) * 2;
      *(unsigned int*)&As[row][ku] =
          *(const unsigned int*)(h + (size_t)(m0 + row) * HID + k0 + ku);
    }
    // ---- stage B (W2 tile, transposed), packed dword stores ----
#pragma unroll
    for (int it = 0; it < 2; ++it) {
      int q  = tid + it * 256;
      int kp = q >> 5;
      int nq = (q & 31) * 4;
      const float* src = W2e + (size_t)(k0 + 2 * kp) * DIM + n0 + nq;
      const float4 fa = *(const float4*)src;
      const float4 fb = *(const float4*)(src + DIM);
      if (k0 + KT < HID) __builtin_prefetch(src + (size_t)KT * DIM, 0, 0);
      *(unsigned int*)&Bs[nq + 0][2 * kp] = pack_bf16(fa.x, fb.x);
      *(unsigned int*)&Bs[nq + 1][2 * kp] = pack_bf16(fa.y, fb.y);
      *(unsigned int*)&Bs[nq + 2][2 * kp] = pack_bf16(fa.z, fb.z);
      *(unsigned int*)&Bs[nq + 3][2 * kp] = pack_bf16(fa.w, fb.w);
    }
    __syncthreads();

    Frag a;
    int arow = wave * 16 + r;
#pragma unroll
    for (int v = 0; v < 8; ++v) {
      int k = (v & 3) * 2 + (v >> 2) * 16 + half * 8;
      a.u[v] = *(const unsigned int*)&As[arow][k];
    }
    Frag bcur, bnxt;
#pragma unroll
    for (int v = 0; v < 8; ++v)
      bcur.u[v] = *(const unsigned int*)&Bs[r][(v & 3) * 2 + (v >> 2) * 16 + half * 8];
#pragma unroll
    for (int nt = 0; nt < 8; ++nt) {
      if (nt < 7) {
        int bcol = (nt + 1) * 16 + r;
#pragma unroll
        for (int v = 0; v < 8; ++v)
          bnxt.u[v] = *(const unsigned int*)&Bs[bcol][(v & 3) * 2 + (v >> 2) * 16 + half * 8];
      }
      acc[nt] = __builtin_amdgcn_wmma_f32_16x16x32_bf16(
          false, a.v, false, bcur.v, (short)0, acc[nt], false, false);
      bcur = bnxt;
    }
    __syncthreads();
  }

  // epilogue: +b2, weight, scatter-add (atomic; cross-expert collisions are in
  // different stream-ordered launches -> deterministic)
#pragma unroll
  for (int nt = 0; nt < 8; ++nt) {
#pragma unroll
    for (int v = 0; v < 8; ++v) {
      int row = m0 + wave * 16 + half * 8 + v;
      int col = n0 + nt * 16 + r;
      float val = acc[nt][v] + b2[e * DIM + col];
      float w   = w_eff[e * CAP + row];
      int   t   = tok[e * CAP + row];
      unsafeAtomicAdd(&y[(size_t)t * DIM + col], val * w);
    }
  }
}

// ---------------- contrib accumulation ----------------
__global__ void k_contrib(const int* __restrict__ tok,
                          const float* __restrict__ w_eff,
                          float* __restrict__ contrib) {
  int i = blockIdx.x * 256 + threadIdx.x;
  if (i < NEXP * CAP) {
    float w = w_eff[i];
    if (w != 0.0f) unsafeAtomicAdd(&contrib[tok[i]], w);
  }
}

// ---------------- passthrough for dropped tokens ----------------
__global__ void k_final(const float* __restrict__ x,
                        const float* __restrict__ contrib,
                        float* __restrict__ y) {
  size_t i = (size_t)blockIdx.x * 256 + threadIdx.x;   // BT*DIM elements
  int t = (int)(i >> 10);                              // DIM = 1024
  if (contrib[t] <= 1e-12f) y[i] = x[i];
}

// ---------------- host-side launch ----------------
extern "C" void kernel_launch(void* const* d_in, const int* in_sizes, int n_in,
                              void* d_out, int out_size, void* d_ws, size_t ws_size,
                              hipStream_t stream) {
  (void)in_sizes; (void)n_in; (void)out_size; (void)ws_size;
  const float* x  = (const float*)d_in[0];
  const float* Wg = (const float*)d_in[1];
  const float* W1 = (const float*)d_in[2];
  const float* b1 = (const float*)d_in[3];
  const float* W2 = (const float*)d_in[4];
  const float* b2 = (const float*)d_in[5];
  // d_in[6] = k (==2, hardcoded)

  float* y    = (float*)d_out;                 // [BT, DIM]
  float* loss = y + (size_t)BT * DIM;
  float* ovf  = loss + 1;

  char* ws = (char*)d_ws;
  float*          assign_w = (float*)(ws + 0);        //  64 KB
  int*            assign_e = (int*)  (ws + 65536);    //  64 KB
  int*            tokb     = (int*)  (ws + 131072);   //  80 KB
  float*          w_eff    = (float*)(ws + 212992);   //  80 KB
  int*            counts   = (int*)  (ws + 294912);   //  32 B
  float*          contrib  = (float*)(ws + 295168);   //  32 KB
  unsigned short* hbuf     = (unsigned short*)(ws + 327936); // 20 MB (CAP x HID bf16)

  // zero-init accumulators / outputs (atomics accumulate; ws poisoned once)
  k_zero<<<(BT * DIM + 255) / 256, 256, 0, stream>>>(y, BT * DIM);
  k_zero<<<(BT + 255) / 256,       256, 0, stream>>>(contrib, BT);
  k_zero<<<1,                      256, 0, stream>>>((float*)counts, NEXP);
  k_zero<<<(NEXP * CAP + 255)/256, 256, 0, stream>>>((float*)tokb, NEXP * CAP);
  k_zero<<<(NEXP * CAP + 255)/256, 256, 0, stream>>>(w_eff, NEXP * CAP);

  k_gate<<<BT / 8, 256, 0, stream>>>(x, Wg, assign_e, assign_w, counts);
  k_select<<<NEXP, 1024, 0, stream>>>(assign_e, assign_w, tokb, w_eff);
  k_losses<<<1, 32, 0, stream>>>(counts, loss, ovf);
  k_contrib<<<(NEXP * CAP + 255) / 256, 256, 0, stream>>>(tokb, w_eff, contrib);

  for (int e = 0; e < NEXP; ++e) {
    k_ffn1<<<dim3(HID / 128, CAP / 128), 256, 0, stream>>>(x, W1, b1, tokb, hbuf, e);
    k_ffn2<<<dim3(DIM / 128, CAP / 128), 256, 0, stream>>>(hbuf, W2, b2, tokb, w_eff, y, e);
  }

  k_final<<<(BT * DIM + 255) / 256, 256, 0, stream>>>(x, contrib, y);
}
